// DecoderRNNTAtt_55722905698592
// MI455X (gfx1250) — compile-verified, matrix operand order checked by reading, not run on hardware
//
#include <hip/hip_runtime.h>
#include <cstddef>

// ---------------------------------------------------------------------------
// RNN-T decoder w/ additive attention for MI455X (gfx1250, wave32, WMMA).
//
// Roofline: joint net = 126 GFLOP GEMM (M=120000,K=512,N=1024) + 491MB fp32
// output (~21us HBM write @23.3TB/s). Joint runs as f16 WMMA
// (v_wmma_f32_16x16x32_f16) with tanh fused into the A-fragment staging in
// LDS; A/B fragments are software-pipelined (double buffered) so ds_load for
// step i+1 overlaps the XDL issue of step i. Serial scan (120 steps, ~5 GFLOP
// total) runs as small fp32 kernels, 3 launches per step.
// ---------------------------------------------------------------------------

#define B_    4
#define T_    250
#define U_    120
#define E_    512     // EMBED == EPROJS == DUNITS == ATT == JOINT == 512
#define ODIM_ 1024

typedef __attribute__((ext_vector_type(16))) _Float16 v16h;
typedef __attribute__((ext_vector_type(4)))  _Float16 v4h;
typedef __attribute__((ext_vector_type(8)))  float    v8f;

// workspace byte offsets
#define OFF_PRE   0u          // pre_enc : B*T*512 f32   (2,048,000 B)
#define OFF_HENC  2048000u    // henc    : B*T*512 f32   (2,048,000 B)
#define OFF_HDEC  4096000u    // hdec    : B*U*512 f32   (  983,040 B)
#define OFF_WP    5079040u    // packed W_lin_out f16    (1,048,576 B)
#define OFF_Z0    6127616u    // 4 x (B*512 f32) state, contiguous
#define OFF_C0    6135808u
#define OFF_Z1    6144000u
#define OFF_C1    6152192u
#define OFF_XBUF  6160384u    // B*1024 f32 (ey | att_c)
#define OFF_G0    6176768u    // B*2048 f32 gates of LSTM0

__device__ __forceinline__ float sigm(float x) { return 1.0f / (1.0f + __expf(-x)); }

// ---------------------------------------------------------------------------
// zero the recurrent state (must happen every launch: determinism)
// ---------------------------------------------------------------------------
__global__ void init_state_kernel(float* st) {
  int i = blockIdx.x * blockDim.x + threadIdx.x;
  if (i < 4 * B_ * 512) st[i] = 0.0f;
}

// ---------------------------------------------------------------------------
// pre_enc[row,a] = hs[row,:] @ W_att_enc[:,a] + b_att_enc[a]
// henc  [row,j] = hs[row,:] . W_lin_enc[j,:] + b_lin_enc[j]
// grid: B*T blocks x 256
// ---------------------------------------------------------------------------
__global__ __launch_bounds__(256)
void prep_proj_kernel(const float* __restrict__ hs,
                      const float* __restrict__ W_att_enc, const float* __restrict__ b_att_enc,
                      const float* __restrict__ W_lin_enc, const float* __restrict__ b_lin_enc,
                      float* __restrict__ pre_enc, float* __restrict__ henc) {
  __shared__ float hsl[512];
  int row = blockIdx.x, tid = threadIdx.x;
  hsl[tid]       = hs[row * 512 + tid];
  hsl[tid + 256] = hs[row * 512 + tid + 256];
  __syncthreads();
  for (int j = tid; j < 512; j += 256) {
    float a0 = b_att_enc[j];
    float h0 = b_lin_enc[j];
    const float* wrow = W_lin_enc + (size_t)j * 512;
    for (int k = 0; k < 512; ++k) {
      a0 += hsl[k] * W_att_enc[k * 512 + j];
      h0 += hsl[k] * wrow[k];
    }
    pre_enc[row * 512 + j] = a0;
    henc[row * 512 + j]    = h0;
  }
}

// ---------------------------------------------------------------------------
// Pack W_lin_out (1024x512 f32, row-major [o][k]) into f16 WMMA B-fragments:
// slot = ((ot*16+kt)*32+lane)*16+e ; N = ot*16+(lane&15)
// K = kt*32 + (e&7) + (e<8?0:16) + (lane<16?0:8)   (16-bit fragment layout)
// ---------------------------------------------------------------------------
__global__ void pack_w_kernel(const float* __restrict__ W, _Float16* __restrict__ Wp) {
  int s = blockIdx.x * blockDim.x + threadIdx.x;
  if (s >= ODIM_ * 512) return;
  int e    = s & 15;
  int lane = (s >> 4) & 31;
  int kt   = (s >> 9) & 15;
  int ot   = s >> 13;
  int o = ot * 16 + (lane & 15);
  int k = kt * 32 + (e & 7) + (e < 8 ? 0 : 16) + (lane < 16 ? 0 : 8);
  Wp[s] = (_Float16)W[(size_t)o * 512 + k];
}

// ---------------------------------------------------------------------------
// per-step S1: additive attention + build x = [embed(y_u) | att_c]
// grid: B blocks x 256 threads
// ---------------------------------------------------------------------------
__global__ __launch_bounds__(256)
void att_step_kernel(const float* __restrict__ hs, const float* __restrict__ pre_enc,
                     const float* __restrict__ W_att_dec, const float* __restrict__ gvec,
                     const int* __restrict__ hlens, const int* __restrict__ ys,
                     const float* __restrict__ embed,
                     const float* __restrict__ z0, float* __restrict__ xbuf, int u) {
  __shared__ float z0l[512];
  __shared__ float ql[512];
  __shared__ float el[256];
  __shared__ float red[256];
  int b = blockIdx.x, tid = threadIdx.x;
  z0l[tid]       = z0[b * 512 + tid];
  z0l[tid + 256] = z0[b * 512 + tid + 256];
  // embedding half of x (independent of attention)
  int y = ys[b * U_ + u];
  xbuf[b * 1024 + tid]       = embed[(size_t)y * 512 + tid];
  xbuf[b * 1024 + tid + 256] = embed[(size_t)y * 512 + tid + 256];
  __syncthreads();
  // q = z0 @ W_att_dec
  for (int a = tid; a < 512; a += 256) {
    float q = 0.0f;
    for (int k = 0; k < 512; ++k) q += z0l[k] * W_att_dec[k * 512 + a];
    ql[a] = q;
  }
  __syncthreads();
  // scores e[t] = tanh(pre_enc + q) . gvec, masked
  int hl = hlens[b];
  float e = -1e9f;
  if (tid < T_ && tid < hl) {
    const float* pe = pre_enc + (size_t)(b * T_ + tid) * 512;
    float s = 0.0f;
    for (int a = 0; a < 512; ++a) s += tanhf(pe[a] + ql[a]) * gvec[a];
    e = s;
  }
  el[tid] = e;
  red[tid] = e;
  __syncthreads();
  for (int st = 128; st > 0; st >>= 1) {
    if (tid < st) red[tid] = fmaxf(red[tid], red[tid + st]);
    __syncthreads();
  }
  float m = red[0];
  __syncthreads();
  float p = __expf(el[tid] - m);
  red[tid] = p;
  __syncthreads();
  for (int st = 128; st > 0; st >>= 1) {
    if (tid < st) red[tid] += red[tid + st];
    __syncthreads();
  }
  float inv = 1.0f / red[0];
  __syncthreads();
  el[tid] = p * inv;  // normalized weights
  __syncthreads();
  // att_c[d] = sum_t w[t] * hs[b,t,d]  -> second half of x
  for (int d = tid; d < 512; d += 256) {
    float acc = 0.0f;
    for (int t = 0; t < T_; ++t) acc += el[t] * hs[(size_t)(b * T_ + t) * 512 + d];
    xbuf[b * 1024 + 512 + d] = acc;
  }
}

// ---------------------------------------------------------------------------
// per-step S2: LSTM0 gates = x@W_ih0.T + b_ih0 + z0@W_hh0.T + b_hh0
// grid: 16 blocks x 512 (4 blocks per batch element)
// ---------------------------------------------------------------------------
__global__ __launch_bounds__(512)
void gates0_kernel(const float* __restrict__ xbuf, const float* __restrict__ z0,
                   const float* __restrict__ W_ih0, const float* __restrict__ b_ih0,
                   const float* __restrict__ W_hh0, const float* __restrict__ b_hh0,
                   float* __restrict__ gates0) {
  __shared__ float xl[1024];
  __shared__ float zl[512];
  int b    = blockIdx.x >> 2;
  int gblk = (blockIdx.x & 3) * 512;
  int tid  = threadIdx.x;
  xl[tid]       = xbuf[b * 1024 + tid];
  xl[tid + 512] = xbuf[b * 1024 + tid + 512];
  zl[tid]       = z0[b * 512 + tid];
  __syncthreads();
  int g = gblk + tid;
  float acc = b_ih0[g] + b_hh0[g];
  const float* wi = W_ih0 + (size_t)g * 1024;
  const float* wh = W_hh0 + (size_t)g * 512;
  for (int k = 0; k < 1024; ++k) acc += xl[k] * wi[k];
  for (int k = 0; k < 512; ++k) acc += zl[k] * wh[k];
  gates0[b * 2048 + g] = acc;
}

// ---------------------------------------------------------------------------
// per-step S3: cell-update0 -> LSTM1 -> cell-update1 -> hdec row
// grid: B blocks x 512 threads (thread k owns hidden unit k)
// ---------------------------------------------------------------------------
__global__ __launch_bounds__(512)
void rest_step_kernel(const float* __restrict__ gates0,
                      const float* __restrict__ W_ih1, const float* __restrict__ b_ih1,
                      const float* __restrict__ W_hh1, const float* __restrict__ b_hh1,
                      const float* __restrict__ W_lin_dec,
                      float* __restrict__ z0, float* __restrict__ c0,
                      float* __restrict__ z1, float* __restrict__ c1,
                      float* __restrict__ hdec, int u) {
  __shared__ float z0l[512];
  __shared__ float z1l[512];
  __shared__ float z1nl[512];
  int b = blockIdx.x, k = threadIdx.x;
  const float* gb = gates0 + b * 2048;
  // LSTM0 cell update (torch gate order i,f,g,o)
  float ig = sigm(gb[k]);
  float fg = sigm(gb[512 + k]);
  float gg = tanhf(gb[1024 + k]);
  float og = sigm(gb[1536 + k]);
  float c0n  = fg * c0[b * 512 + k] + ig * gg;
  float z0nv = og * tanhf(c0n);
  c0[b * 512 + k] = c0n;
  z0[b * 512 + k] = z0nv;
  z0l[k] = z0nv;
  z1l[k] = z1[b * 512 + k];
  __syncthreads();
  // LSTM1 gates for unit k (4 gates of the same unit)
  float gi = b_ih1[k]        + b_hh1[k];
  float gf = b_ih1[512 + k]  + b_hh1[512 + k];
  float gG = b_ih1[1024 + k] + b_hh1[1024 + k];
  float gO = b_ih1[1536 + k] + b_hh1[1536 + k];
  const float* wii = W_ih1 + (size_t)k * 512;
  const float* wif = W_ih1 + (size_t)(512 + k) * 512;
  const float* wig = W_ih1 + (size_t)(1024 + k) * 512;
  const float* wio = W_ih1 + (size_t)(1536 + k) * 512;
  const float* whi = W_hh1 + (size_t)k * 512;
  const float* whf = W_hh1 + (size_t)(512 + k) * 512;
  const float* whg = W_hh1 + (size_t)(1024 + k) * 512;
  const float* who = W_hh1 + (size_t)(1536 + k) * 512;
  for (int kk = 0; kk < 512; ++kk) {
    float a = z0l[kk], h = z1l[kk];
    gi += a * wii[kk] + h * whi[kk];
    gf += a * wif[kk] + h * whf[kk];
    gG += a * wig[kk] + h * whg[kk];
    gO += a * wio[kk] + h * who[kk];
  }
  float c1n  = sigm(gf) * c1[b * 512 + k] + sigm(gi) * tanhf(gG);
  float z1nv = sigm(gO) * tanhf(c1n);
  c1[b * 512 + k] = c1n;
  z1[b * 512 + k] = z1nv;
  z1nl[k] = z1nv;
  __syncthreads();
  // hdec[b,u,k] = z1n . W_lin_dec[k,:]
  float acc = 0.0f;
  const float* wd = W_lin_dec + (size_t)k * 512;
  for (int kk = 0; kk < 512; ++kk) acc += z1nl[kk] * wd[kk];
  hdec[((size_t)b * U_ + u) * 512 + k] = acc;
}

// ---------------------------------------------------------------------------
// Joint network: out[b,t,u,o] = tanh(henc[b,t,:]+hdec[b,u,:]) @ W^T + bias
// One block per (b,t); 512 threads = 16 waves. A (128u x 512k, u padded)
// staged as f16 fragments in 128KB LDS; each wave: 4 o-tiles x 8 u-tiles,
// 16 K-steps of v_wmma_f32_16x16x32_f16. Bias folded into C init.
// A/B fragment loads are double-buffered: load(i+1) issues before wmma(i).
// ---------------------------------------------------------------------------
__global__ __launch_bounds__(512)
void joint_kernel(const float* __restrict__ henc, const float* __restrict__ hdec,
                  const _Float16* __restrict__ Wp, const float* __restrict__ b_lin_out,
                  float* __restrict__ out) {
  extern __shared__ __align__(128) char smem[];
  float*    henc_l = (float*)smem;               // 512 f32  (2KB)
  _Float16* As     = (_Float16*)(smem + 2048);   // 65536 f16 (128KB)
  int bt  = blockIdx.x;              // b*T + t
  int b   = bt / T_;
  int tid = threadIdx.x;
  henc_l[tid & 511] = henc[(size_t)bt * 512 + (tid & 511)];
  __syncthreads();
  // ---- stage A fragments: A[u,k] = f16(tanh(henc[k] + hdec[u,k])) ----
  // 4 contiguous K per slot -> float4 global reads + ds_store_b64 writes.
  const float* hdec_b = hdec + (size_t)b * U_ * 512;
  for (int s = tid; s < 16384; s += 512) {   // 16384 f16-quads
    int eq   = s & 3;                        // e = 4*eq .. 4*eq+3 (same K group)
    int lane = (s >> 2) & 31;
    int frag = s >> 7;                       // frag = ut*16 + kt
    int ut = frag >> 4, kt = frag & 15;
    int u  = ut * 16 + (lane & 15);
    int e0 = eq * 4;
    int k0 = kt * 32 + (e0 & 7) + (e0 < 8 ? 0 : 16) + (lane < 16 ? 0 : 8);
    v4h q = {(_Float16)0.0f, (_Float16)0.0f, (_Float16)0.0f, (_Float16)0.0f};
    if (u < U_) {
      float4 hd = *(const float4*)(hdec_b + u * 512 + k0);
      q.x = (_Float16)tanhf(henc_l[k0]     + hd.x);
      q.y = (_Float16)tanhf(henc_l[k0 + 1] + hd.y);
      q.z = (_Float16)tanhf(henc_l[k0 + 2] + hd.z);
      q.w = (_Float16)tanhf(henc_l[k0 + 3] + hd.w);
    }
    *(v4h*)(As + frag * 512 + lane * 16 + e0) = q;
  }
  __syncthreads();
  // ---- WMMA compute (double-buffered fragment streams) ----
  int wave = tid >> 5;
  int lane = tid & 31;
  int n      = lane & 15;
  int m_base = (lane >> 4) << 3;  // C/D layout: VGPR r -> M = r + 8*(lane>=16)
  const _Float16* As_l = As + lane * 16;
  for (int oi = 0; oi < 4; ++oi) {
    int ot = wave * 4 + oi;
    int o0 = ot * 16;
    const _Float16* wp_base = Wp + ((size_t)ot * 16 * 32 + lane) * 16;
    float bv = b_lin_out[o0 + n];
    v8f acc[8];
#pragma unroll
    for (int ut = 0; ut < 8; ++ut) acc[ut] = (v8f){bv, bv, bv, bv, bv, bv, bv, bv};
    v16h bf_cur = *(const v16h*)(wp_base);             // B(kt=0)
    v16h af_cur = *(const v16h*)(As_l);                // A(kt=0, ut=0)
    for (int kt = 0; kt < 16; ++kt) {
      v16h bf_next = bf_cur;
      if (kt < 15) bf_next = *(const v16h*)(wp_base + (kt + 1) * 512);
#pragma unroll
      for (int ut = 0; ut < 8; ++ut) {
        v16h af_next = af_cur;
        if (ut < 7)       af_next = *(const v16h*)(As_l + ((ut + 1) * 16 + kt) * 512);
        else if (kt < 15) af_next = *(const v16h*)(As_l + (kt + 1) * 512);
        acc[ut] = __builtin_amdgcn_wmma_f32_16x16x32_f16(
            false, af_cur, false, bf_cur, (short)0, acc[ut], false, false);
        af_cur = af_next;
      }
      bf_cur = bf_next;
    }
#pragma unroll
    for (int ut = 0; ut < 8; ++ut) {
#pragma unroll
      for (int r = 0; r < 8; ++r) {
        int u = ut * 16 + m_base + r;
        if (u < U_) out[((size_t)bt * U_ + u) * ODIM_ + o0 + n] = acc[ut][r];
      }
    }
  }
}

// ---------------------------------------------------------------------------
extern "C" void kernel_launch(void* const* d_in, const int* in_sizes, int n_in,
                              void* d_out, int out_size, void* d_ws, size_t ws_size,
                              hipStream_t stream) {
  (void)in_sizes; (void)n_in; (void)out_size; (void)ws_size;
  const float* hs        = (const float*)d_in[0];
  const int*   ys        = (const int*)d_in[1];
  const int*   hlens     = (const int*)d_in[2];
  const float* embed     = (const float*)d_in[3];
  const float* W_ih0     = (const float*)d_in[4];
  const float* b_ih0     = (const float*)d_in[5];
  const float* W_hh0     = (const float*)d_in[6];
  const float* b_hh0     = (const float*)d_in[7];
  const float* W_ih1     = (const float*)d_in[8];
  const float* b_ih1     = (const float*)d_in[9];
  const float* W_hh1     = (const float*)d_in[10];
  const float* b_hh1     = (const float*)d_in[11];
  const float* W_att_enc = (const float*)d_in[12];
  const float* b_att_enc = (const float*)d_in[13];
  const float* W_att_dec = (const float*)d_in[14];
  const float* gvec      = (const float*)d_in[15];
  const float* W_lin_enc = (const float*)d_in[16];
  const float* b_lin_enc = (const float*)d_in[17];
  const float* W_lin_dec = (const float*)d_in[18];
  const float* W_lin_out = (const float*)d_in[19];
  const float* b_lin_out = (const float*)d_in[20];

  char* ws = (char*)d_ws;
  float*    pre_enc = (float*)(ws + OFF_PRE);
  float*    henc    = (float*)(ws + OFF_HENC);
  float*    hdec    = (float*)(ws + OFF_HDEC);
  _Float16* Wp      = (_Float16*)(ws + OFF_WP);
  float*    z0      = (float*)(ws + OFF_Z0);
  float*    c0      = (float*)(ws + OFF_C0);
  float*    z1      = (float*)(ws + OFF_Z1);
  float*    c1      = (float*)(ws + OFF_C1);
  float*    xbuf    = (float*)(ws + OFF_XBUF);
  float*    gates0  = (float*)(ws + OFF_G0);

  // prep (parallel, order-independent among themselves)
  init_state_kernel<<<32, 256, 0, stream>>>(z0);  // zeroes z0,c0,z1,c1 (contiguous)
  prep_proj_kernel<<<B_ * T_, 256, 0, stream>>>(hs, W_att_enc, b_att_enc,
                                                W_lin_enc, b_lin_enc, pre_enc, henc);
  pack_w_kernel<<<(ODIM_ * 512 + 255) / 256, 256, 0, stream>>>(W_lin_out, Wp);

  // sequential decode scan over u
  for (int u = 0; u < U_; ++u) {
    att_step_kernel<<<B_, 256, 0, stream>>>(hs, pre_enc, W_att_dec, gvec, hlens, ys,
                                            embed, z0, xbuf, u);
    gates0_kernel<<<16, 512, 0, stream>>>(xbuf, z0, W_ih0, b_ih0, W_hh0, b_hh0, gates0);
    rest_step_kernel<<<B_, 512, 0, stream>>>(gates0, W_ih1, b_ih1, W_hh1, b_hh1,
                                             W_lin_dec, z0, c0, z1, c1, hdec, u);
  }

  // joint network: the big WMMA GEMM (134 GFLOP incl. padding)
  size_t shmem = 2048 + 65536 * sizeof(_Float16);  // 130 KB (LDS limit: 320 KB)
  joint_kernel<<<B_ * T_, 512, shmem, stream>>>(henc, hdec, Wp, b_lin_out, (float*)d_out);
}
